// MultiheadAttention_87591563034791
// MI455X (gfx1250) — matrix operand order repaired and checked by reading, not compiled, not run on hardware
//
#include <hip/hip_runtime.h>
#include <hip/hip_bf16.h>

// ---------------------------------------------------------------------------
// MHA with RoPE for MI455X (gfx1250): bf16 WMMA everywhere, f32 accumulate.
// T=1024, B=8, E=1280, H=20, D=64.
// Round 3: LDS double-buffering — async global->LDS copies (ASYNCcnt) for the
// next tile overlap with WMMA compute on the current tile; one barrier per
// iteration. DPP16 softmax reductions; packed-b32 V transpose staging.
// ---------------------------------------------------------------------------

typedef __attribute__((ext_vector_type(16))) __bf16 v16bf;
typedef __attribute__((ext_vector_type(8)))  float  v8f;

union Frag {
    uint4 q[2];
    v16bf v;
};

__device__ __forceinline__ unsigned short f2bf(float f) {
    unsigned int u = __float_as_uint(f);
    unsigned int r = (u + 0x7FFFu + ((u >> 16) & 1u)) >> 16;
    return (unsigned short)r;
}
__device__ __forceinline__ float bf2f(unsigned short h) {
    return __uint_as_float(((unsigned int)h) << 16);
}
__device__ __forceinline__ v8f v8f_zero() {
    v8f z;
#pragma unroll
    for (int i = 0; i < 8; ++i) z[i] = 0.0f;
    return z;
}

// ---- DPP16 butterfly reduction over each 16-lane half (all lanes get result)
template <int CTRL>
__device__ __forceinline__ float dppmov(float x) {
    return __int_as_float(
        __builtin_amdgcn_update_dpp(0, __float_as_int(x), CTRL, 0xF, 0xF, true));
}
__device__ __forceinline__ float redmax16(float x) {
    x = fmaxf(x, dppmov<0xB1>(x));   // quad_perm [1,0,3,2]
    x = fmaxf(x, dppmov<0x4E>(x));   // quad_perm [2,3,0,1]
    x = fmaxf(x, dppmov<0x141>(x));  // row_half_mirror
    x = fmaxf(x, dppmov<0x140>(x));  // row_mirror
    return x;
}
__device__ __forceinline__ float redsum16(float x) {
    x += dppmov<0xB1>(x);
    x += dppmov<0x4E>(x);
    x += dppmov<0x141>(x);
    x += dppmov<0x140>(x);
    return x;
}

// ---- Async global->LDS copy (16B per lane), tracked by ASYNCcnt.
// IOFFSET is added to BOTH the LDS and the global address (ISA 15.18.3).
#define ASYNC_CP16(ldsoff, gaddr, OFFSTR)                                  \
    asm volatile("global_load_async_to_lds_b128 %0, %1, off offset:" OFFSTR \
                 ::"v"(ldsoff), "v"(gaddr)                                  \
                 : "memory")
__device__ __forceinline__ void async_wait0() {
    asm volatile("s_wait_asynccnt 0x0" ::: "memory");
}
__device__ __forceinline__ unsigned lds_off(const void* p) {
    return (unsigned)(unsigned long long)p;
}

// ---------------------------------------------------------------------------
// fp32 -> bf16 conversion (4 elems / thread)
// ---------------------------------------------------------------------------
__global__ void cvt_f32_bf16(const float* __restrict__ src,
                             unsigned short* __restrict__ dst, int n) {
    int i = (blockIdx.x * blockDim.x + threadIdx.x) * 4;
    if (i + 3 < n) {
        float4 v = *(const float4*)(src + i);
        union { unsigned short u[4]; uint2 d; } w;
        w.u[0] = f2bf(v.x); w.u[1] = f2bf(v.y);
        w.u[2] = f2bf(v.z); w.u[3] = f2bf(v.w);
        *(uint2*)(dst + i) = w.d;
    }
}

// ---------------------------------------------------------------------------
// Tiled WMMA GEMM:  Y[M,N] = A[M,K] * Wt[N,K]^T + bias
// block = 128 threads (4 waves). WG tile 128(M) x 64(N), K-step 32.
// Double-buffered LDS: async-copy tile k+1 while computing tile k.
// MODE 0: bf16 out scattered to [B,H,T,D];  MODE 1: f32 out row-major [M,N].
// ---------------------------------------------------------------------------
#define GT_M 128
#define GT_N 64
#define GT_K 32

template <int MODE>
__global__ __launch_bounds__(128) void gemm_bf16(
    const unsigned short* __restrict__ A,   // [M,K] bf16
    const unsigned short* __restrict__ Wt,  // [N,K] bf16
    const float* __restrict__ bias,         // [N]
    void* __restrict__ outp, int M, int N, int K) {
    __shared__ unsigned short As[2][GT_M * GT_K];  // [m][k]
    __shared__ unsigned short Bs[2][GT_N * GT_K];  // [n][k]

    const int tid   = threadIdx.x;
    const int wave  = tid >> 5;
    const int lane  = tid & 31;
    const int l16   = lane & 15;
    const int lhalf = lane >> 4;
    const int m0    = blockIdx.y * GT_M;
    const int n0    = blockIdx.x * GT_N;

    v8f acc[2][4];
#pragma unroll
    for (int mi = 0; mi < 2; ++mi)
#pragma unroll
        for (int ni = 0; ni < 4; ++ni) acc[mi][ni] = v8f_zero();

    // Issue async copies for one K-tile into LDS buffer `buf`.
    auto stage = [&](int buf, int koff) {
        unsigned long long ga =
            (unsigned long long)(A + (size_t)(m0 + tid) * K + koff);
        unsigned al = lds_off(&As[buf][tid * GT_K]);  // 64B row per thread
        ASYNC_CP16(al, ga, "0");
        ASYNC_CP16(al, ga, "16");
        ASYNC_CP16(al, ga, "32");
        ASYNC_CP16(al, ga, "48");
        unsigned long long gb = (unsigned long long)(
            Wt + (size_t)(n0 + (tid >> 1)) * K + koff + (tid & 1) * 16);
        unsigned bl = lds_off(&Bs[buf][(tid >> 1) * GT_K + (tid & 1) * 16]);
        ASYNC_CP16(bl, gb, "0");
        ASYNC_CP16(bl, gb, "16");
    };

    // Prologue: stage first tile.
    stage(0, 0);
    async_wait0();
    __syncthreads();

    int cur = 0;
    for (int k0 = 0; k0 < K; k0 += GT_K) {
        const int nxt = cur ^ 1;
        if (k0 + GT_K < K) stage(nxt, k0 + GT_K);  // overlap with compute

        Frag a[2];
#pragma unroll
        for (int mi = 0; mi < 2; ++mi) {
            const unsigned short* ap =
                &As[cur][(wave * 32 + mi * 16 + l16) * GT_K];
            a[mi].q[0] = *(const uint4*)(ap + 16 * lhalf);
            a[mi].q[1] = *(const uint4*)(ap + 16 * lhalf + 8);
        }
#pragma unroll
        for (int ni = 0; ni < 4; ++ni) {
            Frag b;
            const unsigned short* bp =
                &Bs[cur][(ni * 16 + l16) * GT_K + 16 * lhalf];
            b.q[0] = ((const uint4*)bp)[0];
            b.q[1] = ((const uint4*)bp)[1];
#pragma unroll
            for (int mi = 0; mi < 2; ++mi) {
                acc[mi][ni] = __builtin_amdgcn_wmma_f32_16x16x32_bf16(
                    false, a[mi].v, false, b.v, (short)0, acc[mi][ni],
                    false, false);
            }
        }
        async_wait0();
        __syncthreads();
        cur = nxt;
    }

    // Epilogue
#pragma unroll
    for (int ni = 0; ni < 4; ++ni) {
        const int col = n0 + ni * 16 + l16;
        const float bv = bias[col];
#pragma unroll
        for (int mi = 0; mi < 2; ++mi) {
#pragma unroll
            for (int r = 0; r < 8; ++r) {
                const int row = m0 + wave * 32 + mi * 16 + r + 8 * lhalf;
                const float val = acc[mi][ni][r] + bv;
                if (MODE == 0) {
                    const int t = row >> 3, bb = row & 7;
                    const int h = col >> 6, d = col & 63;
                    ((unsigned short*)
                         outp)[(((size_t)bb * 20 + h) * 1024 + t) * 64 + d] =
                        f2bf(val);
                } else {
                    ((float*)outp)[(size_t)row * N + col] = val;
                }
            }
        }
    }
}

// ---------------------------------------------------------------------------
// RoPE applied in-place to Q and K (bf16, [B,H,T,D], rotate-half (d, d+32)).
// ---------------------------------------------------------------------------
__global__ void rope_kernel(unsigned short* __restrict__ Q,
                            unsigned short* __restrict__ Kt) {
    const int i = blockIdx.x * blockDim.x + threadIdx.x;  // < 160*1024*32
    const int d = i & 31;
    const int t = (i >> 5) & 1023;
    const int bh = i >> 15;
    const float ang = (float)t * __expf(-0.28782313662425574f * (float)d);
    float s, c;
    __sincosf(ang, &s, &c);
    const size_t base = ((size_t)bh * 1024 + t) * 64 + d;

    float x1 = bf2f(Q[base]), x2 = bf2f(Q[base + 32]);
    Q[base]      = f2bf(x1 * c - x2 * s);
    Q[base + 32] = f2bf(x2 * c + x1 * s);

    x1 = bf2f(Kt[base]); x2 = bf2f(Kt[base + 32]);
    Kt[base]      = f2bf(x1 * c - x2 * s);
    Kt[base + 32] = f2bf(x2 * c + x1 * s);
}

// ---------------------------------------------------------------------------
// Flash attention per (b,h). Grid: x = T/64 query tiles, y = B*H.
// Block 128 threads = 4 waves; wave owns 16 query rows, full D=64 output.
// Double-buffered K (async->LDS) and V (global->regs early, transpose-store
// after compute); one barrier per 32-key block.
// ---------------------------------------------------------------------------
__global__ __launch_bounds__(128) void flash_kernel(
    const unsigned short* __restrict__ Q,
    const unsigned short* __restrict__ Kg,
    const unsigned short* __restrict__ Vg,
    unsigned short* __restrict__ ctx) {
    __shared__ unsigned short Qs[64 * 64];         // [qrow][dim]
    __shared__ unsigned short Ks[2][32 * 64];      // [key][dim]
    __shared__ unsigned short Vs[2][64 * 32];      // [dim][key] (transposed)
    __shared__ unsigned short Ps[4 * 16 * 32];     // per-wave P tile

    const int tid   = threadIdx.x;
    const int wave  = tid >> 5;
    const int lane  = tid & 31;
    const int l16   = lane & 15;
    const int lhalf = lane >> 4;
    const int bh    = blockIdx.y;  // b*20 + h
    const int q0    = blockIdx.x * 64;
    const size_t headOff = (size_t)bh * 1024 * 64;

    const int vkp = tid & 15;   // key pair
    const int vdc = tid >> 4;   // dim chunk of 8

    auto stageK = [&](int buf, int kb) {
        unsigned long long ga = (unsigned long long)(
            Kg + headOff + (size_t)(kb + (tid >> 2)) * 64 + (tid & 3) * 16);
        unsigned kl = lds_off(&Ks[buf][(tid >> 2) * 64 + (tid & 3) * 16]);
        ASYNC_CP16(kl, ga, "0");
        ASYNC_CP16(kl, ga, "16");
    };
    auto loadV = [&](int kb, uint4& va, uint4& vb) {
        va = *(const uint4*)(Vg + headOff + (size_t)(kb + 2 * vkp) * 64 +
                             vdc * 8);
        vb = *(const uint4*)(Vg + headOff + (size_t)(kb + 2 * vkp + 1) * 64 +
                             vdc * 8);
    };
    auto storeV = [&](int buf, uint4 va, uint4 vb) {
        union { uint4 q; unsigned short u[8]; } ua, ub;
        ua.q = va; ub.q = vb;
#pragma unroll
        for (int e = 0; e < 8; ++e) {
            unsigned pk = (unsigned)ua.u[e] | ((unsigned)ub.u[e] << 16);
            *(unsigned*)&Vs[buf][(vdc * 8 + e) * 32 + 2 * vkp] = pk;
        }
    };

    // Prologue: async-load Q tile + first K tile; stage first V tile.
    {
        const int row = tid >> 1, hf = tid & 1;
        const unsigned qLds = lds_off(Qs + row * 64 + 32 * hf);
        unsigned long long ga = (unsigned long long)(
            Q + headOff + (size_t)(q0 + row) * 64 + 32 * hf);
        ASYNC_CP16(qLds, ga, "0");
        ASYNC_CP16(qLds, ga, "16");
        ASYNC_CP16(qLds, ga, "32");
        ASYNC_CP16(qLds, ga, "48");
    }
    stageK(0, 0);
    {
        uint4 va, vb;
        loadV(0, va, vb);
        storeV(0, va, vb);
    }
    async_wait0();
    __syncthreads();

    v8f o[4];
#pragma unroll
    for (int ni = 0; ni < 4; ++ni) o[ni] = v8f_zero();
    float mrow[8], lrow[8];
#pragma unroll
    for (int r = 0; r < 8; ++r) { mrow[r] = -3.0e38f; lrow[r] = 0.0f; }
    const float scale = 0.125f;  // 64^-0.5

    int cur = 0;
    for (int kb = 0; kb < 1024; kb += 32) {
        const int nxt = cur ^ 1;
        const bool havenext = (kb + 32) < 1024;
        uint4 va, vb;
        if (havenext) {
            stageK(nxt, kb + 32);     // async: overlaps with compute below
            loadV(kb + 32, va, vb);   // vmem latency hidden by S/softmax
        }

        // S = Q * K^T  (16 rows x 32 keys per wave, 2 C-tiles)
        v8f s0 = v8f_zero(), s1 = v8f_zero();
#pragma unroll
        for (int kc = 0; kc < 2; ++kc) {
            Frag aq;
            const unsigned short* ap = Qs + (wave * 16 + l16) * 64 + kc * 32;
            aq.q[0] = *(const uint4*)(ap + 16 * lhalf);
            aq.q[1] = *(const uint4*)(ap + 16 * lhalf + 8);
            Frag bk;
            const unsigned short* bp0 =
                &Ks[cur][l16 * 64 + kc * 32 + 16 * lhalf];
            bk.q[0] = ((const uint4*)bp0)[0];
            bk.q[1] = ((const uint4*)bp0)[1];
            s0 = __builtin_amdgcn_wmma_f32_16x16x32_bf16(
                false, aq.v, false, bk.v, (short)0, s0, false, false);
            const unsigned short* bp1 =
                &Ks[cur][(16 + l16) * 64 + kc * 32 + 16 * lhalf];
            bk.q[0] = ((const uint4*)bp1)[0];
            bk.q[1] = ((const uint4*)bp1)[1];
            s1 = __builtin_amdgcn_wmma_f32_16x16x32_bf16(
                false, aq.v, false, bk.v, (short)0, s1, false, false);
        }

        // Online softmax in C-layout (row = r + 8*lhalf, col = l16),
        // reductions over keys via DPP16 butterflies (pure VALU).
        unsigned short* Pw = Ps + wave * 512;
#pragma unroll
        for (int r = 0; r < 8; ++r) {
            float v0 = s0[r] * scale, v1 = s1[r] * scale;
            float mx = redmax16(fmaxf(v0, v1));
            const float mnew = fmaxf(mrow[r], mx);
            const float p0 = __expf(v0 - mnew);
            const float p1 = __expf(v1 - mnew);
            const float rs = redsum16(p0 + p1);
            const float alpha = __expf(mrow[r] - mnew);
            lrow[r] = lrow[r] * alpha + rs;
            mrow[r] = mnew;
#pragma unroll
            for (int ni = 0; ni < 4; ++ni) o[ni][r] *= alpha;
            const int rl = r + 8 * lhalf;
            Pw[rl * 32 + l16]      = f2bf(p0);
            Pw[rl * 32 + 16 + l16] = f2bf(p1);
        }

        // O += P * V   (P tile is wave-private; same-wave DS ops are in
        // order, so no barrier needed between the stores above and loads).
        Frag aP;
        {
            const unsigned short* pp = Pw + l16 * 32;
            aP.q[0] = *(const uint4*)(pp + 16 * lhalf);
            aP.q[1] = *(const uint4*)(pp + 16 * lhalf + 8);
        }
#pragma unroll
        for (int ni = 0; ni < 4; ++ni) {
            Frag bv;
            const unsigned short* vp =
                &Vs[cur][(ni * 16 + l16) * 32 + 16 * lhalf];
            bv.q[0] = ((const uint4*)vp)[0];
            bv.q[1] = ((const uint4*)vp)[1];
            o[ni] = __builtin_amdgcn_wmma_f32_16x16x32_bf16(
                false, aP.v, false, bv.v, (short)0, o[ni], false, false);
        }

        // Transpose-store next V tile (Vs[nxt] was last read a full
        // iteration ago, so it is safe to overwrite before the barrier).
        if (havenext) storeV(nxt, va, vb);

        async_wait0();
        __syncthreads();
        cur = nxt;
    }

    // Epilogue: normalize and write ctx[t*8+b, h*64+d] (bf16).
    const int h = bh % 20, b = bh / 20;
#pragma unroll
    for (int ni = 0; ni < 4; ++ni) {
#pragma unroll
        for (int r = 0; r < 8; ++r) {
            const int rl = r + 8 * lhalf;
            const int t = q0 + wave * 16 + rl;
            const float val = o[ni][r] / lrow[r];
            const size_t off =
                (size_t)(t * 8 + b) * 1280 + h * 64 + ni * 16 + l16;
            ctx[off] = f2bf(val);
        }
    }
}

// ---------------------------------------------------------------------------
// Host-side orchestration.
// ---------------------------------------------------------------------------
extern "C" void kernel_launch(void* const* d_in, const int* in_sizes, int n_in,
                              void* d_out, int out_size, void* d_ws,
                              size_t ws_size, hipStream_t stream) {
    const float* query = (const float*)d_in[0];
    const float* q_w = (const float*)d_in[1];
    const float* q_b = (const float*)d_in[2];
    const float* k_w = (const float*)d_in[3];
    const float* k_b = (const float*)d_in[4];
    const float* v_w = (const float*)d_in[5];
    const float* v_b = (const float*)d_in[6];
    const float* out_w = (const float*)d_in[7];
    const float* out_b = (const float*)d_in[8];

    const int M = 8192;  // T*B
    const int E = 1280;
    char* ws = (char*)d_ws;
    unsigned short* Xbf = (unsigned short*)(ws + 0);         // 8192x1280
    unsigned short* WQ  = (unsigned short*)(ws + 20971520);  // 1280x1280
    unsigned short* WK  = (unsigned short*)(ws + 24248320);
    unsigned short* WV  = (unsigned short*)(ws + 27525120);
    unsigned short* WO  = (unsigned short*)(ws + 30801920);
    unsigned short* Qb  = (unsigned short*)(ws + 34078720);  // [B,H,T,D]
    unsigned short* Kb  = (unsigned short*)(ws + 55050240);
    unsigned short* Vb  = (unsigned short*)(ws + 76021760);
    unsigned short* CTX = (unsigned short*)(ws + 96993280);  // 8192x1280

    // 1) fp32 -> bf16 conversions
    cvt_f32_bf16<<<(M * E) / 1024, 256, 0, stream>>>(query, Xbf, M * E);
    cvt_f32_bf16<<<(E * E) / 1024, 256, 0, stream>>>(q_w, WQ, E * E);
    cvt_f32_bf16<<<(E * E) / 1024, 256, 0, stream>>>(k_w, WK, E * E);
    cvt_f32_bf16<<<(E * E) / 1024, 256, 0, stream>>>(v_w, WV, E * E);
    cvt_f32_bf16<<<(E * E) / 1024, 256, 0, stream>>>(out_w, WO, E * E);

    // 2) QKV projections (WMMA), writing [B,H,T,D] bf16
    dim3 gg(E / GT_N, M / GT_M);
    gemm_bf16<0><<<gg, 128, 0, stream>>>(Xbf, WQ, q_b, Qb, M, E, E);
    gemm_bf16<0><<<gg, 128, 0, stream>>>(Xbf, WK, k_b, Kb, M, E, E);
    gemm_bf16<0><<<gg, 128, 0, stream>>>(Xbf, WV, v_b, Vb, M, E, E);

    // 3) RoPE on Q, K
    rope_kernel<<<(160 * 1024 * 32) / 256, 256, 0, stream>>>(Qb, Kb);

    // 4) Flash attention -> CTX [8192,1280] bf16
    flash_kernel<<<dim3(1024 / 64, 160), 128, 0, stream>>>(Qb, Kb, Vb, CTX);

    // 5) Output projection -> d_out f32 [T,B,E]
    gemm_bf16<1><<<gg, 128, 0, stream>>>(CTX, WO, out_b, (float*)d_out, M, E,
                                         E);
}
